// MultiHeadAttentionLayer_87900800680357
// MI455X (gfx1250) — matrix operand order repaired and checked by reading, not compiled, not run on hardware
//
#include <hip/hip_runtime.h>

// MI455X / gfx1250: wave32, WMMA 16x16x32 bf16 -> f32.
typedef __attribute__((ext_vector_type(16))) __bf16    v16bf;
typedef __attribute__((ext_vector_type(8)))  float     v8f;
typedef __attribute__((ext_vector_type(4)))  unsigned  u32x4;
typedef __attribute__((ext_vector_type(2)))  unsigned  u32x2;
typedef __attribute__((ext_vector_type(4)))  float     f32x4;

#define BSZ    2
#define SEQ    2048
#define DM     512
#define HN     8
#define DEP    64
#define MTOT   (BSZ * SEQ)

__device__ __forceinline__ unsigned short f2bf(float f) {
  unsigned u = __float_as_uint(f);
  unsigned r = 0x7FFFu + ((u >> 16) & 1u);
  return (unsigned short)((u + r) >> 16);
}
__device__ __forceinline__ unsigned pk2(float a, float b) {
  return (unsigned)f2bf(a) | ((unsigned)f2bf(b) << 16);
}

union Frag { u32x4 u[2]; v16bf v; };

__device__ __forceinline__ v8f wmma_bf16(const Frag& A, const Frag& B, v8f C) {
  // (neg_a, A, neg_b, B, c_mod, C, reuse_a, reuse_b)
  return __builtin_amdgcn_wmma_f32_16x16x32_bf16(false, A.v, false, B.v,
                                                 (short)0, C, false, false);
}

// ---------------------------------------------------------------------------
// GEMM + bias: Out = X[4096,512] @ W[512,512] + bias
// mode 0: Out bf16 in [B,H,S,64]    (Q, K)
// mode 1: Out bf16 in [B,H,64,S]    (V transposed, for PV B-fragments)
// mode 2: Out f32  in [4096,512]    (final output projection, NT store)
// Block: 128 threads (4 waves). Block tile 64(M) x 64(N), K-steps of 32.
// ---------------------------------------------------------------------------
template <bool A_BF16>
__global__ __launch_bounds__(128)
void proj_kernel(const void* __restrict__ Xv, const float* __restrict__ W,
                 const float* __restrict__ bias, void* __restrict__ Outv,
                 int mode) {
  __shared__ unsigned short As[64 * 40];   // [m][k], stride 40 halves
  __shared__ unsigned short Bs[64 * 40];   // [n][k], stride 40 halves

  const int tid  = threadIdx.x;
  const int lane = tid & 31;
  const int wave = tid >> 5;
  const int m0   = blockIdx.x * 64;
  const int n0   = blockIdx.y * 64;

  v8f z = {0.f, 0.f, 0.f, 0.f, 0.f, 0.f, 0.f, 0.f};
  v8f acc[4] = {z, z, z, z};

  const int arow  = tid >> 1;          // 0..63
  const int ahalf = (tid & 1) * 16;    // 0 or 16
  const int bk    = tid >> 2;          // 0..31
  const int bq    = (tid & 3) * 16;    // 0,16,32,48

  for (int k0 = 0; k0 < DM; k0 += 32) {
    // stage A tile (64 x 32) -> As
    if (A_BF16) {
      const unsigned short* src =
          (const unsigned short*)Xv + (size_t)(m0 + arow) * DM + k0 + ahalf;
      *(u32x4*)(As + arow * 40 + ahalf)     = *(const u32x4*)(src);
      *(u32x4*)(As + arow * 40 + ahalf + 8) = *(const u32x4*)(src + 8);
    } else {
      const float* src = (const float*)Xv + (size_t)(m0 + arow) * DM + k0 + ahalf;
      for (int j = 0; j < 4; ++j) {
        f32x4 f = *(const f32x4*)(src + 4 * j);
        u32x2 p = {pk2(f.x, f.y), pk2(f.z, f.w)};
        *(u32x2*)(As + arow * 40 + ahalf + 4 * j) = p;
      }
    }
    // stage B tile: W rows k0..k0+31, cols n0..n0+63 -> Bs[n][k] (transposed)
    {
      const float* src = W + (size_t)(k0 + bk) * DM + n0 + bq;
      for (int j = 0; j < 4; ++j) {
        f32x4 f = *(const f32x4*)(src + 4 * j);
        Bs[(bq + 4 * j + 0) * 40 + bk] = f2bf(f.x);
        Bs[(bq + 4 * j + 1) * 40 + bk] = f2bf(f.y);
        Bs[(bq + 4 * j + 2) * 40 + bk] = f2bf(f.z);
        Bs[(bq + 4 * j + 3) * 40 + bk] = f2bf(f.w);
      }
    }
    __syncthreads();

    // A fragment: wave row strip 16*wave .. +15
    Frag a;
    {
      const int row = 16 * wave + (lane & 15);
      const int kb  = (lane < 16) ? 0 : 8;          // K pairs: 0..7 / 8..15
      a.u[0] = *(const u32x4*)(As + row * 40 + kb);
      a.u[1] = *(const u32x4*)(As + row * 40 + kb + 16);  // K 16..23 / 24..31
    }
    for (int j = 0; j < 4; ++j) {
      Frag b;
      const int n   = 16 * j + (lane & 15);
      const int kb2 = (lane < 16) ? 0 : 8;          // K 0..7,16..23 / 8..15,24..31
      b.u[0] = *(const u32x4*)(Bs + n * 40 + kb2);
      b.u[1] = *(const u32x4*)(Bs + n * 40 + kb2 + 16);
      acc[j] = wmma_bf16(a, b, acc[j]);
    }
    __syncthreads();
  }

  // epilogue: C layout — VGPR v: M = v (lanes 0-15) / v+8 (lanes 16-31), N = lane&15
  for (int j = 0; j < 4; ++j) {
    const int n    = n0 + 16 * j + (lane & 15);
    const float bv = bias[n];
    for (int v = 0; v < 8; ++v) {
      const int m     = m0 + 16 * wave + ((lane < 16) ? v : v + 8);
      const float val = acc[j][v] + bv;
      if (mode == 2) {
        // final output: write-once stream, keep it out of L2's hot set
        __builtin_nontemporal_store(val, (float*)Outv + (size_t)m * DM + n);
      } else {
        const int b = m >> 11, s = m & (SEQ - 1);
        const int h = n >> 6,  d = n & 63;
        unsigned short* Ob = (unsigned short*)Outv;
        if (mode == 0)
          Ob[((size_t)(b * HN + h) * SEQ + s) * DEP + d] = f2bf(val);
        else
          Ob[((size_t)(b * HN + h) * DEP + d) * SEQ + s] = f2bf(val);
      }
    }
  }
}

// ---------------------------------------------------------------------------
// Attention core: one block per (b, h, 16-query tile). 128 threads / 4 waves.
// scores strip 16 x 2048 f32 held in LDS (~131 KB; CDNA5 WGP has 320 KB,
// so 2 blocks/WGP). attn output (268 MB > 192 MB L2) uses NT stores so the
// one-shot stream doesn't evict the K/V rows feeding the WMMA pipes.
// ---------------------------------------------------------------------------
__global__ __launch_bounds__(128)
void attn_kernel(const unsigned short* __restrict__ Qh,
                 const unsigned short* __restrict__ Kh,
                 const unsigned short* __restrict__ Vt,
                 const float* __restrict__ mask,
                 float* __restrict__ attn,
                 unsigned short* __restrict__ Ctx) {
  __shared__ float sc[16 * 2052];            // [m][key], stride 2052 (pad 4)
  __shared__ unsigned short qs[16 * 72];     // Q tile [m][d], stride 72

  const int tid  = threadIdx.x;
  const int lane = tid & 31;
  const int wave = tid >> 5;
  const int q0   = blockIdx.x * 16;
  const int h    = blockIdx.y;
  const int b    = blockIdx.z;
  const int bh   = b * HN + h;

  // stage Q tile (16 x 64 bf16)
  {
    const int r = tid >> 3, seg = (tid & 7) * 8;
    const unsigned short* src = Qh + ((size_t)bh * SEQ + q0 + r) * DEP + seg;
    *(u32x4*)(qs + r * 72 + seg) = *(const u32x4*)src;
  }
  __syncthreads();

  // Q fragments for d=0..31 and d=32..63
  Frag a0, a1;
  {
    const int row = lane & 15;
    const int kb  = (lane < 16) ? 0 : 8;
    a0.u[0] = *(const u32x4*)(qs + row * 72 + kb);
    a0.u[1] = *(const u32x4*)(qs + row * 72 + kb + 16);
    a1.u[0] = *(const u32x4*)(qs + row * 72 + 32 + kb);
    a1.u[1] = *(const u32x4*)(qs + row * 72 + 32 + kb + 16);
  }

  v8f z = {0.f, 0.f, 0.f, 0.f, 0.f, 0.f, 0.f, 0.f};

  // scores = Q K^T / 8 + mask*(-1e9): wave covers 512 keys (32 tiles of 16)
  for (int it = 0; it < 32; ++it) {
    const int n0k = wave * 512 + it * 16;
    const int key = n0k + (lane & 15);
    const int kb2 = (lane < 16) ? 0 : 8;
    const unsigned short* kp = Kh + ((size_t)bh * SEQ + key) * DEP;
    if (it < 31)  // prefetch next 16-key tile (global_prefetch_b8)
      __builtin_prefetch(kp + 16 * DEP, 0, 3);
    Frag b0, b1;
    b0.u[0] = *(const u32x4*)(kp + kb2);          // d 0..7,16..23 / 8..15,24..31
    b0.u[1] = *(const u32x4*)(kp + kb2 + 16);
    b1.u[0] = *(const u32x4*)(kp + 32 + kb2);     // d 32..39,48..55 / 40..47,56..63
    b1.u[1] = *(const u32x4*)(kp + 32 + kb2 + 16);
    v8f c = z;
    c = wmma_bf16(a0, b0, c);
    c = wmma_bf16(a1, b1, c);
    const float mv = mask[b * SEQ + key] * -1e9f;
    for (int v = 0; v < 8; ++v) {
      const int m = (lane < 16) ? v : v + 8;
      sc[m * 2052 + key] = c[v] * 0.125f + mv;
    }
  }
  __syncthreads();

  // softmax: row r = tid/8, 8 lanes per row, shfl_xor tree within wave32
  {
    const int r = tid >> 3, j = tid & 7;
    float* rowp = sc + r * 2052 + j * 256;
    float mx = -3.4e38f;
    for (int i = 0; i < 64; ++i) {
      f32x4 f = *(const f32x4*)(rowp + 4 * i);
      mx = fmaxf(mx, fmaxf(fmaxf(f.x, f.y), fmaxf(f.z, f.w)));
    }
    for (int off = 1; off < 8; off <<= 1) mx = fmaxf(mx, __shfl_xor(mx, off));
    float sum = 0.f;
    for (int i = 0; i < 64; ++i) {
      f32x4 f = *(const f32x4*)(rowp + 4 * i);
      sum += __expf(f.x - mx) + __expf(f.y - mx) +
             __expf(f.z - mx) + __expf(f.w - mx);
    }
    for (int off = 1; off < 8; off <<= 1) sum += __shfl_xor(sum, off);
    const float inv = 1.f / sum;
    float* dst = attn + ((size_t)bh * SEQ + q0 + r) * SEQ + j * 256;
    for (int i = 0; i < 64; ++i) {
      f32x4 f = *(const f32x4*)(rowp + 4 * i);
      f.x = __expf(f.x - mx) * inv;
      f.y = __expf(f.y - mx) * inv;
      f.z = __expf(f.z - mx) * inv;
      f.w = __expf(f.w - mx) * inv;
      *(f32x4*)(rowp + 4 * i) = f;                       // normalized probs for PV
      __builtin_nontemporal_store(f, (f32x4*)(dst + 4 * i)); // 268 MB stream, TH=NT
    }
  }
  __syncthreads();

  // PV: wave computes d-tile 16*wave..+15; B from transposed V (contiguous)
  v8f acc = z;
  const int dn = 16 * wave + (lane & 15);
  const unsigned short* vp = Vt + ((size_t)bh * DEP + dn) * SEQ;
  for (int kk = 0; kk < SEQ; kk += 32) {
    Frag a;
    {
      const int row = lane & 15;
      const float* sp = sc + row * 2052 + kk + ((lane < 16) ? 0 : 8);
      f32x4 f0 = *(const f32x4*)(sp);
      f32x4 f1 = *(const f32x4*)(sp + 4);
      f32x4 f2 = *(const f32x4*)(sp + 16);
      f32x4 f3 = *(const f32x4*)(sp + 20);
      a.u[0] = (u32x4){pk2(f0.x, f0.y), pk2(f0.z, f0.w),
                       pk2(f1.x, f1.y), pk2(f1.z, f1.w)};
      a.u[1] = (u32x4){pk2(f2.x, f2.y), pk2(f2.z, f2.w),
                       pk2(f3.x, f3.y), pk2(f3.z, f3.w)};
    }
    Frag bfr;
    const int kb2 = kk + ((lane < 16) ? 0 : 8);
    bfr.u[0] = *(const u32x4*)(vp + kb2);
    bfr.u[1] = *(const u32x4*)(vp + kb2 + 16);
    acc = wmma_bf16(a, bfr, acc);
  }
  for (int v = 0; v < 8; ++v) {
    const int m = (lane < 16) ? v : v + 8;
    const int s = q0 + m;
    Ctx[((size_t)(b * SEQ + s)) * DM + h * DEP + dn] = f2bf(acc[v]);
  }
}

// ---------------------------------------------------------------------------
extern "C" void kernel_launch(void* const* d_in, const int* in_sizes, int n_in,
                              void* d_out, int out_size, void* d_ws,
                              size_t ws_size, hipStream_t stream) {
  const float* query = (const float*)d_in[0];
  const float* key_  = (const float*)d_in[1];
  const float* value = (const float*)d_in[2];
  const float* mask  = (const float*)d_in[3];
  const float* Wq = (const float*)d_in[4];
  const float* bq = (const float*)d_in[5];
  const float* Wk = (const float*)d_in[6];
  const float* bk = (const float*)d_in[7];
  const float* Wv = (const float*)d_in[8];
  const float* bv = (const float*)d_in[9];
  const float* Wo = (const float*)d_in[10];
  const float* bo = (const float*)d_in[11];

  char* ws = (char*)d_ws;
  const size_t HALF_MAT = (size_t)MTOT * DM * sizeof(unsigned short); // 4 MB
  unsigned short* Qh  = (unsigned short*)(ws);
  unsigned short* Kh  = (unsigned short*)(ws + HALF_MAT);
  unsigned short* Vt  = (unsigned short*)(ws + 2 * HALF_MAT);
  unsigned short* Ctx = (unsigned short*)(ws + 3 * HALF_MAT);

  float* out  = (float*)d_out;
  float* attn = out + (size_t)BSZ * SEQ * DM;

  dim3 gp(MTOT / 64, DM / 64, 1);   // 64 x 8 blocks
  dim3 bp(128);

  proj_kernel<false><<<gp, bp, 0, stream>>>((const void*)query, Wq, bq, (void*)Qh, 0);
  proj_kernel<false><<<gp, bp, 0, stream>>>((const void*)key_,  Wk, bk, (void*)Kh, 0);
  proj_kernel<false><<<gp, bp, 0, stream>>>((const void*)value, Wv, bv, (void*)Vt, 1);

  attn_kernel<<<dim3(SEQ / 16, HN, BSZ), bp, 0, stream>>>(Qh, Kh, Vt, mask, attn, Ctx);

  proj_kernel<true><<<gp, bp, 0, stream>>>((const void*)Ctx, Wo, bo, (void*)out, 2);
}
